// GINBase_21225728377481
// MI455X (gfx1250) — matrix operand order repaired and checked by reading, not compiled, not run on hardware
//
#include <hip/hip_runtime.h>
#include <hip/hip_bf16.h>

// ---------------- CDNA5 WMMA types ----------------
typedef __attribute__((ext_vector_type(16))) __bf16 v16bf;
typedef __attribute__((ext_vector_type(8)))  float  v8f;

#define DIM 64
#define LN_EPS 1e-5f

__device__ __forceinline__ v8f wmma_bf16(v16bf a, v16bf b, v8f c) {
  return __builtin_amdgcn_wmma_f32_16x16x32_bf16(
      /*neg_a=*/false, a, /*neg_b=*/false, b,
      /*c_mod=*/(short)0, c, /*reuse_a=*/false, /*reuse_b=*/false);
}

// B fragment from pre-swizzled weights: tile (kt, nt), each lane's 16 halfs are
// contiguous 32B -> one global b128 pair. ntiles = Ncols/16.
__device__ __forceinline__ v16bf frag_b_sw(const __bf16* w, int ntiles, int kt, int nt,
                                           int lane) {
  return ((const v16bf*)w)[((size_t)kt * ntiles + nt) * 32 + lane];
}

// A fragment with fused LayerNorm + ReLU from f32 LDS tile s[16][ld].
// Layout: lane L<16: h=0..7 -> K=0..7, h=8..15 -> K=16..23 ; L>=16: +8.
__device__ __forceinline__ v16bf frag_a_ln(const float* s, int ld, int k0, int lane,
                                           const float* rm, const float* rs,
                                           const float* g, const float* b) {
  int M  = lane & 15;
  int kl = (lane & 16) ? 8 : 0;
  float m = rm[M], r = rs[M];
  v16bf a;
#pragma unroll
  for (int h = 0; h < 16; ++h) {
    int K = (h & 7) + ((h & 8) << 1) + kl + k0;
    float v = (s[M * ld + K] - m) * r * g[K] + b[K];
    a[h] = (__bf16)fmaxf(v, 0.0f);
  }
  return a;
}

// Scatter one float4 (row r, cols c..c+3 of a 16-row tile) into bf16
// A-fragment-layout LDS: frag[kt][lane][h], 4 consecutive halfs per store.
__device__ __forceinline__ void put_a4(__bf16* fragbuf, int r, int c,
                                       float x0, float x1, float x2, float x3) {
  int kt = c >> 5, k = c & 31;
  int ln = r + (((k >> 3) & 1) << 4);
  int h0 = (k & 7) + ((k & 16) ? 8 : 0);
  __bf16* p = fragbuf + ((kt * 32 + ln) << 4) + h0;
  p[0] = (__bf16)x0; p[1] = (__bf16)x1; p[2] = (__bf16)x2; p[3] = (__bf16)x3;
}

// Store v8f accumulator tile (+bias[col]) to f32 LDS s[16][ld] at column block n0.
__device__ __forceinline__ void store_d(float* s, int ld, int n0, int lane, v8f acc,
                                        const float* bias) {
  int col = n0 + (lane & 15);
  int mb  = (lane & 16) ? 8 : 0;
  float bv = bias[col];
#pragma unroll
  for (int r = 0; r < 8; ++r)
    s[(mb + r) * ld + col] = acc[r] + bv;
}

// ---------------- helper kernels ----------------
__global__ void k_zero4(float* __restrict__ p, int n4) {
  int i = blockIdx.x * blockDim.x + threadIdx.x;
  if (i < n4) ((float4*)p)[i] = make_float4(0.f, 0.f, 0.f, 0.f);
}

// f32 weights [K][Nc] -> bf16 fragment-swizzled: out[((kt*ntiles+nt)*32+lane)*16+h]
__global__ void k_swz_bf16(const float* __restrict__ w, __bf16* __restrict__ out,
                           int Nc, int n) {
  int i = blockIdx.x * blockDim.x + threadIdx.x;
  if (i >= n) return;
  int h    = i & 15;
  int lane = (i >> 4) & 31;
  int tile = i >> 9;
  int ntiles = Nc >> 4;
  int nt = tile % ntiles;
  int kt = tile / ntiles;
  int Ki = kt * 32 + ((lane & 16) ? 8 : 0) + (h & 7) + ((h & 8) << 1);
  int Ni = nt * 16 + (lane & 15);
  out[i] = (__bf16)w[(size_t)Ki * Nc + Ni];
}

// msg = relu(x[src] + e) scatter-added into agg[dst]. One thread = 4 floats of one edge.
__global__ void k_edge_message(const float* __restrict__ x, const float* __restrict__ ea,
                               const int* __restrict__ src, const int* __restrict__ dst,
                               float* __restrict__ agg, int E) {
  int i = blockIdx.x * blockDim.x + threadIdx.x;
  int e = i >> 4;
  if (e >= E) return;
  int c = (i & 15) * 4;
  int s = src[e], d = dst[e];
  float4 xv = *(const float4*)(x + (size_t)s * DIM + c);
  float4 ev = *(const float4*)(ea + (size_t)e * DIM + c);
  float* a = agg + (size_t)d * DIM + c;
  atomicAdd(a + 0, fmaxf(xv.x + ev.x, 0.f));
  atomicAdd(a + 1, fmaxf(xv.y + ev.y, 0.f));
  atomicAdd(a + 2, fmaxf(xv.z + ev.z, 0.f));
  atomicAdd(a + 3, fmaxf(xv.w + ev.w, 0.f));
}

// ---------------- node MLP: 16 nodes / block, 4 waves ----------------
__global__ __launch_bounds__(128)
void k_node_mlp(float* __restrict__ x, const float* __restrict__ agg,
                const __bf16* __restrict__ w1, const float* __restrict__ b1,
                const float* __restrict__ g1, const float* __restrict__ be1,
                const __bf16* __restrict__ w2, const float* __restrict__ b2,
                const float* __restrict__ epsp,
                const float* __restrict__ ng, const float* __restrict__ nb,
                int doRelu) {
  __shared__ alignas(32) __bf16 hinbf[2 * 32 * 16];  // GEMM1 A tiles, frag layout
  __shared__ float h1[16 * 128];
  __shared__ float h2[16 * 64];
  __shared__ float rm[16], rs[16];
  int t = threadIdx.x, lane = t & 31, wave = t >> 5;
  int row0 = blockIdx.x * 16;
  float epl = 1.0f + epsp[0];

  // hin = (1+eps)*x + agg  -> bf16 fragment layout in LDS
#pragma unroll
  for (int i = 0; i < 2; ++i) {
    int id = t + i * 128;
    int r = id >> 4, c = (id & 15) * 4;
    size_t g = (size_t)(row0 + r) * DIM + c;
    float4 xv = *(const float4*)(x + g);
    float4 av = *(const float4*)(agg + g);
    put_a4(hinbf, r, c, epl * xv.x + av.x, epl * xv.y + av.y,
                         epl * xv.z + av.z, epl * xv.w + av.w);
  }
  __syncthreads();

  // GEMM1: [16x64] @ [64x128]  (2 N-tiles / wave, 2 K-tiles)
  v8f acc[2] = {};
#pragma unroll
  for (int kt = 0; kt < 2; ++kt) {
    v16bf a = ((const v16bf*)hinbf)[kt * 32 + lane];
#pragma unroll
    for (int j = 0; j < 2; ++j) {
      v16bf b = frag_b_sw(w1, 8, kt, wave * 2 + j, lane);
      acc[j] = wmma_bf16(a, b, acc[j]);
    }
  }
#pragma unroll
  for (int j = 0; j < 2; ++j) store_d(h1, 128, (wave * 2 + j) * 16, lane, acc[j], b1);
  __syncthreads();

  if (t < 16) {  // LN stats over 128
    float s = 0.f, s2 = 0.f;
    for (int c = 0; c < 128; ++c) { float v = h1[t * 128 + c]; s += v; s2 += v * v; }
    float m = s * (1.0f / 128.0f);
    rm[t] = m; rs[t] = rsqrtf(s2 * (1.0f / 128.0f) - m * m + LN_EPS);
  }
  __syncthreads();

  // GEMM2: relu(LN(h1)) @ [128x64]  (1 N-tile / wave, 4 K-tiles)
  v8f acc2 = {};
  int n0 = wave * 16;
#pragma unroll
  for (int kt = 0; kt < 4; ++kt) {
    v16bf a = frag_a_ln(h1, 128, kt * 32, lane, rm, rs, g1, be1);
    v16bf b = frag_b_sw(w2, 4, kt, wave, lane);
    acc2 = wmma_bf16(a, b, acc2);
  }
  store_d(h2, 64, n0, lane, acc2, b2);
  __syncthreads();

  if (t < 16) {  // LN stats over 64
    float s = 0.f, s2 = 0.f;
    for (int c = 0; c < 64; ++c) { float v = h2[t * 64 + c]; s += v; s2 += v * v; }
    float m = s * (1.0f / 64.0f);
    rm[t] = m; rs[t] = rsqrtf(s2 * (1.0f / 64.0f) - m * m + LN_EPS);
  }
  __syncthreads();

  // x = (relu?)(LN(h2)*ng+nb) + x
#pragma unroll
  for (int i = 0; i < 8; ++i) {
    int id = t + i * 128;
    int r = id >> 6, c = id & 63;
    float v = (h2[r * 64 + c] - rm[r]) * rs[r] * ng[c] + nb[c];
    if (doRelu) v = fmaxf(v, 0.f);
    size_t g = (size_t)(row0 + r) * DIM + c;
    x[g] = v + x[g];
  }
}

// ---------------- edge update: 16 edges / block, 4 waves ----------------
__global__ __launch_bounds__(128)
void k_edge_update(float* __restrict__ ea, const float* __restrict__ x,
                   const int* __restrict__ src, const int* __restrict__ dst,
                   const __bf16* __restrict__ ew1, const float* __restrict__ eb1,
                   const float* __restrict__ eg, const float* __restrict__ ebt,
                   const __bf16* __restrict__ ew2, const float* __restrict__ eb2) {
  __shared__ alignas(32) __bf16 zsbf[6 * 32 * 16];  // GEMM1 A tiles, frag layout
  __shared__ float h1[16 * 192];
  __shared__ float h2[16 * 64];
  __shared__ float rm[16], rs[16];
  __shared__ int sidx[16], didx[16];
  int t = threadIdx.x, lane = t & 31, wave = t >> 5;
  int e0 = blockIdx.x * 16;
  if (t < 16) { sidx[t] = src[e0 + t]; didx[t] = dst[e0 + t]; }
  __syncthreads();

  // z = concat(x[src], x[dst], edge_attr) -> bf16 fragment layout in LDS
#pragma unroll
  for (int i = 0; i < 6; ++i) {
    int id = t + i * 128;
    int r = id / 48, c = (id % 48) * 4;
    float4 v;
    if (c < 64)       v = *(const float4*)(x + (size_t)sidx[r] * DIM + c);
    else if (c < 128) v = *(const float4*)(x + (size_t)didx[r] * DIM + (c - 64));
    else              v = *(const float4*)(ea + (size_t)(e0 + r) * DIM + (c - 128));
    put_a4(zsbf, r, c, v.x, v.y, v.z, v.w);
  }
  __syncthreads();

  // GEMM1: [16x192] @ [192x192]  (3 N-tiles / wave, 6 K-tiles)
  v8f acc[3] = {};
#pragma unroll
  for (int kt = 0; kt < 6; ++kt) {
    v16bf a = ((const v16bf*)zsbf)[kt * 32 + lane];
#pragma unroll
    for (int j = 0; j < 3; ++j) {
      v16bf b = frag_b_sw(ew1, 12, kt, wave * 3 + j, lane);
      acc[j] = wmma_bf16(a, b, acc[j]);
    }
  }
#pragma unroll
  for (int j = 0; j < 3; ++j) store_d(h1, 192, (wave * 3 + j) * 16, lane, acc[j], eb1);
  __syncthreads();

  if (t < 16) {  // LN stats over 192
    float s = 0.f, s2 = 0.f;
    for (int c = 0; c < 192; ++c) { float v = h1[t * 192 + c]; s += v; s2 += v * v; }
    float m = s * (1.0f / 192.0f);
    rm[t] = m; rs[t] = rsqrtf(s2 * (1.0f / 192.0f) - m * m + LN_EPS);
  }
  __syncthreads();

  // GEMM2: relu(LN(h1)) @ [192x64]  (1 N-tile / wave, 6 K-tiles)
  v8f acc2 = {};
  int n0 = wave * 16;
#pragma unroll
  for (int kt = 0; kt < 6; ++kt) {
    v16bf a = frag_a_ln(h1, 192, kt * 32, lane, rm, rs, eg, ebt);
    v16bf b = frag_b_sw(ew2, 4, kt, wave, lane);
    acc2 = wmma_bf16(a, b, acc2);
  }
  store_d(h2, 64, n0, lane, acc2, eb2);
  __syncthreads();

  // edge_attr += h2  (residual)
#pragma unroll
  for (int i = 0; i < 8; ++i) {
    int id = t + i * 128;
    int r = id >> 6, c = id & 63;
    size_t g = (size_t)(e0 + r) * DIM + c;
    ea[g] = h2[r * 64 + c] + ea[g];
  }
}

// ---------------- host launcher ----------------
extern "C" void kernel_launch(void* const* d_in, const int* in_sizes, int n_in,
                              void* d_out, int out_size, void* d_ws, size_t ws_size,
                              hipStream_t stream) {
  const int N = 100000, E = 800000, L = 4;
  const float* x_in  = (const float*)d_in[0];
  const float* ea_in = (const float*)d_in[1];
  const int*   ei    = (const int*)d_in[2];
  const int* src = ei;
  const int* dst = ei + E;

  char* ws = (char*)d_ws;
  const size_t XB = (size_t)N * DIM * sizeof(float);  // 25.6 MB
  const size_t EB = (size_t)E * DIM * sizeof(float);  // 204.8 MB
  float*  xbuf = (float*)ws;
  float*  ebuf = (float*)(ws + XB);
  float*  agg  = (float*)(ws + XB + EB);
  __bf16* wbf  = (__bf16*)(ws + XB + EB + XB);        // 4 * 131 KB swizzled bf16 weights

  hipMemcpyAsync(xbuf, x_in, XB, hipMemcpyDeviceToDevice, stream);
  hipMemcpyAsync(ebuf, ea_in, EB, hipMemcpyDeviceToDevice, stream);

  // Swizzle weights to fragment-ordered bf16 (layout per layer: w1 | w2 | ew1 | ew2)
  for (int l = 0; l < L; ++l) {
    const float* w1  = (const float*)d_in[3 + l * 15 + 0];
    const float* w2  = (const float*)d_in[3 + l * 15 + 4];
    const float* ew1 = (const float*)d_in[3 + l * 15 + 9];
    const float* ew2 = (const float*)d_in[3 + l * 15 + 13];
    __bf16* base = wbf + (size_t)l * 65536;
    k_swz_bf16<<<(8192 + 255) / 256, 256, 0, stream>>>(w1, base, 128, 8192);
    k_swz_bf16<<<(8192 + 255) / 256, 256, 0, stream>>>(w2, base + 8192, 64, 8192);
    k_swz_bf16<<<(36864 + 255) / 256, 256, 0, stream>>>(ew1, base + 16384, 192, 36864);
    k_swz_bf16<<<(12288 + 255) / 256, 256, 0, stream>>>(ew2, base + 53248, 64, 12288);
  }

  for (int l = 0; l < L; ++l) {
    const float* b1   = (const float*)d_in[3 + l * 15 + 1];
    const float* g1   = (const float*)d_in[3 + l * 15 + 2];
    const float* be1  = (const float*)d_in[3 + l * 15 + 3];
    const float* b2   = (const float*)d_in[3 + l * 15 + 5];
    const float* eps  = (const float*)d_in[3 + l * 15 + 6];
    const float* ng   = (const float*)d_in[3 + l * 15 + 7];
    const float* nb   = (const float*)d_in[3 + l * 15 + 8];
    const float* eb1  = (const float*)d_in[3 + l * 15 + 10];
    const float* egp  = (const float*)d_in[3 + l * 15 + 11];
    const float* ebt  = (const float*)d_in[3 + l * 15 + 12];
    const float* eb2  = (const float*)d_in[3 + l * 15 + 14];
    __bf16* base = wbf + (size_t)l * 65536;

    int n4 = N * DIM / 4;
    k_zero4<<<(n4 + 255) / 256, 256, 0, stream>>>(agg, n4);
    k_edge_message<<<(E * 16 + 255) / 256, 256, 0, stream>>>(xbuf, ebuf, src, dst, agg, E);
    k_node_mlp<<<N / 16, 128, 0, stream>>>(xbuf, agg,
                                           base, b1, g1, be1,
                                           base + 8192, b2, eps, ng, nb,
                                           (l < L - 1) ? 1 : 0);
    k_edge_update<<<E / 16, 128, 0, stream>>>(ebuf, xbuf, src, dst,
                                              base + 16384, eb1, egp, ebt,
                                              base + 53248, eb2);
  }

  hipMemcpyAsync(d_out, xbuf, XB, hipMemcpyDeviceToDevice, stream);
  hipMemcpyAsync((char*)d_out + XB, ebuf, EB, hipMemcpyDeviceToDevice, stream);
  (void)in_sizes; (void)n_in; (void)out_size; (void)ws_size;
}
